// EpsilonPointNet2_41970420418078
// MI455X (gfx1250) — compile-verified, matrix operand order checked by reading, not run on hardware
//
#include <hip/hip_runtime.h>
#include <hip/hip_bf16.h>

typedef __attribute__((ext_vector_type(16))) _Float16 v16h;
typedef __attribute__((ext_vector_type(8)))  float    v8f;

// ---------------------------------------------------------------------------
// Farthest point sampling: one block per batch, iterative argmax reduction.
// ---------------------------------------------------------------------------
__global__ void fps_kernel(const float* __restrict__ xyz, int n, int npoint,
                           int* __restrict__ out_idx, float* __restrict__ dist) {
    int b = blockIdx.x;
    xyz     += (size_t)b * n * 3;
    dist    += (size_t)b * n;
    out_idx += (size_t)b * npoint;
    const int BS = 256;
    int tid = threadIdx.x;
    __shared__ float sd[BS];
    __shared__ int   si[BS];
    __shared__ int   s_last;
    for (int i = tid; i < n; i += BS) dist[i] = 1e10f;
    if (tid == 0) { out_idx[0] = 0; s_last = 0; }
    __syncthreads();
    for (int it = 1; it < npoint; ++it) {
        int last = s_last;
        float lx = xyz[last*3+0], ly = xyz[last*3+1], lz = xyz[last*3+2];
        float best = -1.0f; int bi = 0;
        for (int i = tid; i < n; i += BS) {
            float dx = xyz[i*3+0]-lx, dy = xyz[i*3+1]-ly, dz = xyz[i*3+2]-lz;
            float d  = dx*dx + dy*dy + dz*dz;
            float nd = fminf(dist[i], d);
            dist[i] = nd;
            if (nd > best) { best = nd; bi = i; }
        }
        sd[tid] = best; si[tid] = bi;
        __syncthreads();
        for (int off = BS/2; off > 0; off >>= 1) {
            if (tid < off) {
                if (sd[tid+off] > sd[tid] ||
                    (sd[tid+off] == sd[tid] && si[tid+off] < si[tid])) {
                    sd[tid] = sd[tid+off]; si[tid] = si[tid+off];
                }
            }
            __syncthreads();
        }
        if (tid == 0) { s_last = si[0]; out_idx[it] = si[0]; }
        __syncthreads();
    }
}

__global__ void gather_xyz_kernel(const float* __restrict__ xyz, const int* __restrict__ idx,
                                  float* __restrict__ out, int B, int n, int S) {
    int tid = blockIdx.x * blockDim.x + threadIdx.x;
    if (tid >= B * S) return;
    int b = tid / S;
    int k = idx[tid];
    const float* src = xyz + ((size_t)b * n + k) * 3;
    float* dst = out + (size_t)tid * 3;
    dst[0] = src[0]; dst[1] = src[1]; dst[2] = src[2];
}

// ---------------------------------------------------------------------------
// Ball query: first `ns` in-radius points in index order, pad with first hit.
// ---------------------------------------------------------------------------
__global__ void ball_query_kernel(const float* __restrict__ xyz, const float* __restrict__ centers,
                                  int B, int n, int S, float r2, int ns, int* __restrict__ idx) {
    int tid = blockIdx.x * blockDim.x + threadIdx.x;
    if (tid >= B * S) return;
    int b = tid / S;
    const float* c = centers + (size_t)tid * 3;
    const float* p = xyz + (size_t)b * n * 3;
    int* o = idx + (size_t)tid * ns;
    float cx = c[0], cy = c[1], cz = c[2];
    int cnt = 0;
    for (int k = 0; k < n && cnt < ns; ++k) {
        float dx = p[k*3+0]-cx, dy = p[k*3+1]-cy, dz = p[k*3+2]-cz;
        if (dx*dx + dy*dy + dz*dz <= r2) {
            if (cnt == 0) { for (int j = 0; j < ns; ++j) o[j] = k; }
            o[cnt++] = k;
        }
    }
    if (cnt == 0) { for (int j = 0; j < ns; ++j) o[j] = 0; }
}

// ---------------------------------------------------------------------------
// Grouping: X[row, 0:3] = xyz[idx]-center ; X[row, 3:] = feats[idx]
// ---------------------------------------------------------------------------
__global__ void group_kernel(const float* __restrict__ xyz, const float* __restrict__ feats,
                             const float* __restrict__ centers, const int* __restrict__ idx,
                             float* __restrict__ X, int B, int n, int S, int ns, int cfeat) {
    int row = blockIdx.x * blockDim.x + threadIdx.x;
    int total = B * S * ns;
    if (row >= total) return;
    int bs = row / ns;
    int b  = bs / S;
    int k  = idx[row];
    const float* p = xyz + ((size_t)b * n + k) * 3;
    const float* c = centers + (size_t)bs * 3;
    int C = 3 + cfeat;
    float* o = X + (size_t)row * C;
    o[0] = p[0] - c[0]; o[1] = p[1] - c[1]; o[2] = p[2] - c[2];
    if (cfeat) {
        const float* f = feats + ((size_t)b * n + k) * cfeat;
        for (int cc = 0; cc < cfeat; ++cc) o[3 + cc] = f[cc];
    }
}

// ---------------------------------------------------------------------------
// WMMA GEMM, LDS-free: Y[M,N] = act(X[M,K] @ W[N,K]^T + bias).
// One wave32 computes a 32x32 tile (2 A-frags x 2 B-frags -> 4 WMMAs / K-step).
// Fragment for v_wmma_f32_16x16x32_f16: lane holds row (lane&15); element e maps
// to K = kk + (lane>=16 ? 8:0) + (e<8 ? e : e+8)  == two contiguous 8-float runs,
// loaded directly from global as float4 pairs and converted to f16 in registers.
// The interior-tile test is hoisted so the hot loop is branch-free.
// act: 0 = none, 1 = relu, 2 = silu
// ---------------------------------------------------------------------------
__device__ __forceinline__ v16h frag_load_vec(const float* __restrict__ row, int k0) {
    const float4* p0 = (const float4*)(row + k0);
    const float4* p1 = (const float4*)(row + k0 + 16);
    float4 x0 = p0[0], x1 = p0[1], x2 = p1[0], x3 = p1[1];
    v16h f;
    f[0]  = (_Float16)x0.x; f[1]  = (_Float16)x0.y;
    f[2]  = (_Float16)x0.z; f[3]  = (_Float16)x0.w;
    f[4]  = (_Float16)x1.x; f[5]  = (_Float16)x1.y;
    f[6]  = (_Float16)x1.z; f[7]  = (_Float16)x1.w;
    f[8]  = (_Float16)x2.x; f[9]  = (_Float16)x2.y;
    f[10] = (_Float16)x2.z; f[11] = (_Float16)x2.w;
    f[12] = (_Float16)x3.x; f[13] = (_Float16)x3.y;
    f[14] = (_Float16)x3.z; f[15] = (_Float16)x3.w;
    return f;
}

__device__ __forceinline__ v16h frag_load_edge(const float* __restrict__ row, int k0, int K) {
    v16h f = {};
    if (row == nullptr) return f;
    for (int e = 0; e < 16; ++e) {
        int k = k0 + (e < 8 ? e : e + 8);
        f[e] = (k < K) ? (_Float16)row[k] : (_Float16)0.0f;
    }
    return f;
}

#define WMMA_F16(A, Bf, C) \
    __builtin_amdgcn_wmma_f32_16x16x32_f16(false, (A), false, (Bf), (short)0, (C), false, false)

__global__ void __launch_bounds__(32)
gemm_bias_act_wmma(const float* __restrict__ X, const float* __restrict__ W,
                   const float* __restrict__ bias, float* __restrict__ Y,
                   int M, int N, int K, int act) {
    int lane = threadIdx.x;
    int m0 = blockIdx.x * 32;
    int n0 = blockIdx.y * 32;
    int lm  = lane & 15;
    int khi = (lane >> 4) * 8;

    int ra0 = m0 + lm, ra1 = m0 + 16 + lm;
    int rb0 = n0 + lm, rb1 = n0 + 16 + lm;
    const float* pa0 = (ra0 < M) ? X + (size_t)ra0 * K : nullptr;
    const float* pa1 = (ra1 < M) ? X + (size_t)ra1 * K : nullptr;
    const float* pb0 = (rb0 < N) ? W + (size_t)rb0 * K : nullptr;
    const float* pb1 = (rb1 < N) ? W + (size_t)rb1 * K : nullptr;

    v8f acc00 = {}, acc01 = {}, acc10 = {}, acc11 = {};

    // uniform across the wave: full 32x32 tile, 16B-aligned rows
    bool interior = ((K & 3) == 0) && (m0 + 32 <= M) && (n0 + 32 <= N);
    int  K32 = K & ~31;

    if (interior) {
        for (int kk = 0; kk < K32; kk += 32) {
            int k0 = kk + khi;
            v16h a0 = frag_load_vec(pa0, k0);
            v16h a1 = frag_load_vec(pa1, k0);
            v16h b0 = frag_load_vec(pb0, k0);
            v16h b1 = frag_load_vec(pb1, k0);
            if (kk + 32 < K32) __builtin_prefetch(pa0 + kk + 32 + khi, 0, 3);
            acc00 = WMMA_F16(a0, b0, acc00);
            acc01 = WMMA_F16(a0, b1, acc01);
            acc10 = WMMA_F16(a1, b0, acc10);
            acc11 = WMMA_F16(a1, b1, acc11);
        }
        if (K32 < K) {
            int k0 = K32 + khi;
            v16h a0 = frag_load_edge(pa0, k0, K);
            v16h a1 = frag_load_edge(pa1, k0, K);
            v16h b0 = frag_load_edge(pb0, k0, K);
            v16h b1 = frag_load_edge(pb1, k0, K);
            acc00 = WMMA_F16(a0, b0, acc00);
            acc01 = WMMA_F16(a0, b1, acc01);
            acc10 = WMMA_F16(a1, b0, acc10);
            acc11 = WMMA_F16(a1, b1, acc11);
        }
    } else {
        for (int kk = 0; kk < K; kk += 32) {
            int k0 = kk + khi;
            v16h a0 = frag_load_edge(pa0, k0, K);
            v16h a1 = frag_load_edge(pa1, k0, K);
            v16h b0 = frag_load_edge(pb0, k0, K);
            v16h b1 = frag_load_edge(pb1, k0, K);
            acc00 = WMMA_F16(a0, b0, acc00);
            acc01 = WMMA_F16(a0, b1, acc01);
            acc10 = WMMA_F16(a1, b0, acc10);
            acc11 = WMMA_F16(a1, b1, acc11);
        }
    }

    // C/D layout: VGPR r -> M = r + (lane>=16 ? 8 : 0); N = lane & 15
    int mb  = m0 + ((lane >> 4) * 8);
    int nc0 = n0 + lm;
    int nc1 = n0 + 16 + lm;
    auto store_tile = [&](const v8f& acc, int mbase, int ncol) {
        if (ncol >= N) return;
        float bv = bias[ncol];
        for (int r = 0; r < 8; ++r) {
            int m = mbase + r;
            if (m < M) {
                float v = acc[r] + bv;
                if (act == 1)      v = fmaxf(v, 0.0f);
                else if (act == 2) v = v / (1.0f + expf(-v));
                Y[(size_t)m * N + ncol] = v;
            }
        }
    };
    store_tile(acc00, mb,      nc0);
    store_tile(acc01, mb,      nc1);
    store_tile(acc10, mb + 16, nc0);
    store_tile(acc11, mb + 16, nc1);
}

// ---------------------------------------------------------------------------
// Max-pool over group dim, write into channel-offset slice of concat output.
// ---------------------------------------------------------------------------
__global__ void maxpool_kernel(const float* __restrict__ Y, float* __restrict__ out,
                               int BS, int ns, int C, int Ctot, int coff) {
    int tid = blockIdx.x * blockDim.x + threadIdx.x;
    if (tid >= BS * C) return;
    int bs = tid / C, c = tid % C;
    const float* p = Y + ((size_t)bs * ns) * C + c;
    float m = p[0];
    for (int j = 1; j < ns; ++j) m = fmaxf(m, p[(size_t)j * C]);
    out[(size_t)bs * Ctot + coff + c] = m;
}

// ---------------------------------------------------------------------------
// 3-NN inverse-squared-distance interpolation -> first Ck channels of X.
// ---------------------------------------------------------------------------
__global__ void three_interp_kernel(const float* __restrict__ ux, const float* __restrict__ kx,
                                    const float* __restrict__ kf, float* __restrict__ X,
                                    int B, int n, int m, int Ck, int Cin) {
    int tid = blockIdx.x * blockDim.x + threadIdx.x;
    if (tid >= B * n) return;
    int b = tid / n;
    const float* u  = ux + (size_t)tid * 3;
    const float* kp = kx + (size_t)b * m * 3;
    float px = u[0], py = u[1], pz = u[2];
    float d0 = 1e30f, d1 = 1e30f, d2 = 1e30f;
    int   i0 = 0,     i1 = 0,     i2 = 0;
    for (int k = 0; k < m; ++k) {
        float dx = kp[k*3+0]-px, dy = kp[k*3+1]-py, dz = kp[k*3+2]-pz;
        float d = dx*dx + dy*dy + dz*dz;
        if (d < d0)      { d2 = d1; i2 = i1; d1 = d0; i1 = i0; d0 = d; i0 = k; }
        else if (d < d1) { d2 = d1; i2 = i1; d1 = d;  i1 = k; }
        else if (d < d2) { d2 = d;  i2 = k; }
    }
    float w0 = 1.0f / (d0 + 1e-8f), w1 = 1.0f / (d1 + 1e-8f), w2 = 1.0f / (d2 + 1e-8f);
    float ws = w0 + w1 + w2; w0 /= ws; w1 /= ws; w2 /= ws;
    const float* f0 = kf + ((size_t)b * m + i0) * Ck;
    const float* f1 = kf + ((size_t)b * m + i1) * Ck;
    const float* f2 = kf + ((size_t)b * m + i2) * Ck;
    float* o = X + (size_t)tid * Cin;
    for (int c = 0; c < Ck; ++c) o[c] = f0[c]*w0 + f1[c]*w1 + f2[c]*w2;
}

__global__ void copy_skip_kernel(const float* __restrict__ uf, float* __restrict__ X,
                                 int rows, int Cu, int Cin, int coff) {
    int tid = blockIdx.x * blockDim.x + threadIdx.x;
    if (tid >= rows * Cu) return;
    int r = tid / Cu, c = tid % Cu;
    X[(size_t)r * Cin + coff + c] = uf[tid];
}

// e = [sin(t*10^linspace(0,3,64)), cos(...)]
__global__ void time_embed_kernel(const float* __restrict__ t, float* __restrict__ e,
                                  int B, int half) {
    int tid = blockIdx.x * blockDim.x + threadIdx.x;
    if (tid >= B * half) return;
    int b = tid / half, j = tid % half;
    float fr = powf(10.0f, 3.0f * (float)j / (float)(half - 1));
    float a = t[b] * fr;
    e[(size_t)b * 2 * half + j]        = sinf(a);
    e[(size_t)b * 2 * half + half + j] = cosf(a);
}

__global__ void concat_h_kernel(const float* __restrict__ f, const float* __restrict__ te,
                                float* __restrict__ h, int rows, int Nper) {
    int tid = blockIdx.x * blockDim.x + threadIdx.x;
    if (tid >= rows * 256) return;
    int r = tid / 256, c = tid % 256;
    int b = r / Nper;
    h[tid] = (c < 128) ? f[(size_t)r * 128 + c] : te[(size_t)b * 128 + (c - 128)];
}

// ---------------------------------------------------------------------------
extern "C" void kernel_launch(void* const* d_in, const int* in_sizes, int n_in_args,
                              void* d_out, int out_size, void* d_ws, size_t ws_size,
                              hipStream_t stream) {
    (void)in_sizes; (void)n_in_args; (void)out_size; (void)ws_size;
    const int B = 4, N = 16384;
    const float* xyz0 = (const float*)d_in[0];
    const float* tvec = (const float*)d_in[1];
    auto P = [&](int i) -> const float* { return (const float*)d_in[i]; };
    // JAX pytree flatten order: dict keys alphabetical -> fc, fp, head, sa, tproj;
    // each linear {"w","b"} -> b, w.
    auto FC_B = [](int l){ return 2 + 2*l; };
    auto FC_W = [](int l){ return 3 + 2*l; };
    auto FP_B = [](int j, int l){ return 6 + j*4 + l*2; };
    auto FP_W = [](int j, int l){ return 7 + j*4 + l*2; };
    auto HD_B = [](int l){ return 22 + 2*l; };
    auto HD_W = [](int l){ return 23 + 2*l; };
    auto SA_B = [](int c, int s, int l){ return 28 + c*12 + s*6 + l*2; };
    auto SA_W = [](int c, int s, int l){ return 29 + c*12 + s*6 + l*2; };
    auto TP_B = [](int l){ return 76 + 2*l; };
    auto TP_W = [](int l){ return 77 + 2*l; };

    char* wsp = (char*)d_ws;
    auto alloc = [&](size_t bytes) -> char* {
        char* p = wsp;
        wsp += (bytes + 255) & ~(size_t)255;
        return p;
    };
    int Sn[5] = { N, 1024, 256, 64, 16 };
    float* xyzl[5]; xyzl[0] = nullptr;
    for (int l = 1; l <= 4; ++l) xyzl[l] = (float*)alloc((size_t)B * Sn[l] * 3 * 4);
    float* fpsdist = (float*)alloc((size_t)B * N * 4);
    int*   fpsidx  = (int*)alloc((size_t)B * 1024 * 4);
    int*   bqidx   = (int*)alloc((size_t)B * 1024 * 32 * 4);
    int FeatC[5] = { 0, 96, 256, 512, 1024 };
    float* featl[5]; featl[0] = nullptr;
    for (int l = 1; l <= 4; ++l) featl[l] = (float*)alloc((size_t)B * Sn[l] * FeatC[l] * 4);
    int FPC[4] = { 128, 256, 512, 512 };   // FP output channels per level
    float* fpout[4];
    for (int i = 0; i < 4; ++i) fpout[i] = (float*)alloc((size_t)B * Sn[i] * FPC[i] * 4);
    float* temb   = (float*)alloc((size_t)B * 128 * 4);
    float* tprojh = (float*)alloc((size_t)B * 128 * 4);
    float* tprojo = (float*)alloc((size_t)B * 128 * 4);
    float* Xbuf   = (float*)alloc((size_t)B * N * 256 * 4);
    float* Ybuf   = (float*)alloc((size_t)B * N * 256 * 4);

    auto cdiv = [](int a, int b){ return (a + b - 1) / b; };
    auto gemm = [&](const float* Xp, int wi, int bi, float* Yp, int M, int Nn, int K, int act) {
        dim3 g((unsigned)((M + 31) / 32), (unsigned)((Nn + 31) / 32));
        gemm_bias_act_wmma<<<g, 32, 0, stream>>>(Xp, P(wi), P(bi), Yp, M, Nn, K, act);
    };

    // ---------------- Set-abstraction (MSG) levels ----------------
    const float radii[4][2]     = {{0.05f,0.1f},{0.1f,0.2f},{0.2f,0.4f},{0.4f,0.8f}};
    const int   nsamp[4][2]     = {{16,32},{16,32},{16,32},{16,32}};
    const int   mdims[4][2][4]  = {
        {{3,16,16,32},      {3,32,32,64}},
        {{99,64,64,128},    {99,64,96,128}},
        {{259,128,196,256}, {259,128,196,256}},
        {{515,256,256,512}, {515,256,384,512}}};
    const int coffs[4][2] = {{0,32},{0,128},{0,256},{0,512}};

    const float* cur_xyz  = xyz0;
    const float* cur_feat = nullptr;
    for (int lv = 0; lv < 4; ++lv) {
        int nin   = Sn[lv];
        int S     = Sn[lv+1];
        int cfeat = (lv == 0) ? 0 : FeatC[lv];
        fps_kernel<<<B, 256, 0, stream>>>(cur_xyz, nin, S, fpsidx, fpsdist);
        gather_xyz_kernel<<<cdiv(B*S,256), 256, 0, stream>>>(cur_xyz, fpsidx, xyzl[lv+1], B, nin, S);
        for (int s = 0; s < 2; ++s) {
            int ns = nsamp[lv][s];
            float r = radii[lv][s];
            ball_query_kernel<<<cdiv(B*S,256), 256, 0, stream>>>(
                cur_xyz, xyzl[lv+1], B, nin, S, r*r, ns, bqidx);
            int rows = B * S * ns;
            group_kernel<<<cdiv(rows,256), 256, 0, stream>>>(
                cur_xyz, cur_feat, xyzl[lv+1], bqidx, Xbuf, B, nin, S, ns, cfeat);
            const int* d = mdims[lv][s];
            gemm(Xbuf, SA_W(lv,s,0), SA_B(lv,s,0), Ybuf, rows, d[1], d[0], 1);
            gemm(Ybuf, SA_W(lv,s,1), SA_B(lv,s,1), Xbuf, rows, d[2], d[1], 1);
            gemm(Xbuf, SA_W(lv,s,2), SA_B(lv,s,2), Ybuf, rows, d[3], d[2], 1);
            maxpool_kernel<<<cdiv(B*S*d[3],256), 256, 0, stream>>>(
                Ybuf, featl[lv+1], B*S, ns, d[3], FeatC[lv+1], coffs[lv][s]);
        }
        cur_xyz  = xyzl[lv+1];
        cur_feat = featl[lv+1];
    }

    // ---------------- Feature propagation, level 3 -> 0 ----------------
    for (int i = 3; i >= 0; --i) {
        int n = Sn[i], m = Sn[i+1];
        const float* ux = (i == 0) ? xyz0 : xyzl[i];
        const float* kx = xyzl[i+1];
        const float* kf = (i == 3) ? featl[4] : fpout[i+1];
        int Ck = (i == 3) ? FeatC[4] : FPC[i+1];
        const float* uf = (i == 0) ? nullptr : featl[i];
        int Cu = (i == 0) ? 0 : FeatC[i];
        int Cin = Ck + Cu;                              // 1536 / 768 / 608 / 256
        three_interp_kernel<<<cdiv(B*n,256), 256, 0, stream>>>(
            ux, kx, kf, Xbuf, B, n, m, Ck, Cin);
        if (uf)
            copy_skip_kernel<<<cdiv(B*n*Cu,256), 256, 0, stream>>>(uf, Xbuf, B*n, Cu, Cin, Ck);
        int d1 = (i >= 2) ? 512 : ((i == 1) ? 256 : 128);
        gemm(Xbuf, FP_W(i,0), FP_B(i,0), Ybuf,     B*n, d1,     Cin, 1);
        gemm(Ybuf, FP_W(i,1), FP_B(i,1), fpout[i], B*n, FPC[i], d1,  1);
    }

    // ---------------- Heads ----------------
    int rowsN = B * N;
    gemm(fpout[0], FC_W(0), FC_B(0), Ybuf, rowsN, 128, 128, 1);   // relu
    gemm(Ybuf,     FC_W(1), FC_B(1), Xbuf, rowsN, 128, 128, 0);   // f -> Xbuf

    time_embed_kernel<<<cdiv(B*64,256), 256, 0, stream>>>(tvec, temb, B, 64);
    gemm(temb,   TP_W(0), TP_B(0), tprojh, B, 128, 128, 2);       // silu
    gemm(tprojh, TP_W(1), TP_B(1), tprojo, B, 128, 128, 0);

    concat_h_kernel<<<cdiv(rowsN*256,256), 256, 0, stream>>>(Xbuf, tprojo, Ybuf, rowsN, N);
    gemm(Ybuf, HD_W(0), HD_B(0), Xbuf, rowsN, 256, 256, 2);       // silu
    gemm(Xbuf, HD_W(1), HD_B(1), Ybuf, rowsN, 128, 256, 2);       // silu
    gemm(Ybuf, HD_W(2), HD_B(2), (float*)d_out, rowsN, 3, 128, 0);
}
